// LiveSR_15401752724120
// MI455X (gfx1250) — matrix operand order repaired and checked by reading, not compiled, not run on hardware
//
#include <hip/hip_runtime.h>
#include <hip/hip_bf16.h>
#include <stdint.h>

// ---------------------------------------------------------------------------
// LiveSR (NAS-MDSR F=48 x4) as implicit-GEMM conv3x3 via CDNA5 WMMA f16->f32.
// E=10 experts, N=128 samples, per-sample expert selected by labels[].
// D[O x Npix] = W[O x K] * im2col(X)[K x Npix], K = (ky*3+kx)*Cpad + c.
// Activations HWC f16 in workspace. A-tile staged to LDS via TDM
// (tensor_load_to_lds) and B halo slab via async global->LDS when available.
// ---------------------------------------------------------------------------

typedef _Float16 half8  __attribute__((ext_vector_type(8)));
typedef _Float16 half16 __attribute__((ext_vector_type(16)));
typedef float    float8 __attribute__((ext_vector_type(8)));
typedef unsigned int u32x4 __attribute__((ext_vector_type(4)));
typedef int          i32x4 __attribute__((ext_vector_type(4)));
typedef int          i32x8 __attribute__((ext_vector_type(8)));

#define SHUF16(lo, hi) __builtin_shufflevector(lo, hi, 0,1,2,3,4,5,6,7,8,9,10,11,12,13,14,15)

#if __has_builtin(__builtin_amdgcn_global_load_async_to_lds_b128) && \
    __has_builtin(__builtin_amdgcn_s_wait_asynccnt)
#define USE_ASYNC_LDS 1
#else
#define USE_ASYNC_LDS 0
#endif

#if __has_builtin(__builtin_amdgcn_tensor_load_to_lds) && \
    __has_builtin(__builtin_amdgcn_s_wait_tensorcnt)
#define USE_TDM 1
#else
#define USE_TDM 0
#endif

typedef __attribute__((address_space(1))) i32x4 as1_v4i;
typedef __attribute__((address_space(3))) i32x4 as3_v4i;

// global VA == addrspace(1) address; int round-trip avoids const/AS friction
static __device__ __forceinline__ as1_v4i* as1p(const void* p) {
    return (as1_v4i*)(unsigned long long)(uintptr_t)p;
}
// generic -> AS3 void -> AS3 v4i (addrspacecast then bitcast)
static __device__ __forceinline__ as3_v4i* as3p(void* p) {
    return (as3_v4i*)(__attribute__((address_space(3))) void*)p;
}

static __device__ __forceinline__ unsigned lds_off_of(const void* p) {
    return (unsigned)(uintptr_t)(__attribute__((address_space(3))) const void*)p;
}

static __device__ __forceinline__ half16 h16_zero() {
    half16 z;
#pragma unroll
    for (int i = 0; i < 16; ++i) z[i] = (_Float16)0.0f;
    return z;
}

#if USE_TDM
// TDM: 2D tile load, rows x kp f16 elements, contiguous rows (stride = kp).
static __device__ __forceinline__ void tdm_load_tile(
    const _Float16* gsrc, unsigned lds_byte_off, int kp, int rows)
{
    unsigned long long ga = (unsigned long long)(uintptr_t)gsrc;
    u32x4 g0;
    g0[0] = 1u;                                        // count=1, user mode
    g0[1] = lds_byte_off;                              // lds_addr
    g0[2] = (unsigned)(ga & 0xffffffffu);              // global_addr lo
    g0[3] = (unsigned)((ga >> 32) & 0x1ffffffu) | 0x80000000u; // hi | type=2
    i32x8 g1;
    g1[0] = 1 << 16;                 // data_size=2B, no multicast/pad/iterate
    g1[1] = (kp & 0xffff) << 16;     // tensor_dim0 lo16 (abar_addr=0)
    g1[2] = (kp >> 16) | (rows << 16); // tensor_dim0 hi | tensor_dim1 lo
    g1[3] = kp << 16;                // tensor_dim1 hi(0) | tile_dim0
    g1[4] = rows;                    // tile_dim1 | tile_dim2=0
    g1[5] = kp;                      // tensor_dim0_stride lo32
    g1[6] = 0;                       // stride hi16 | tensor_dim1_stride lo16
    g1[7] = 0;
    i32x4 z4 = {0, 0, 0, 0};
#if __clang_major__ >= 23
    i32x8 z8 = {0, 0, 0, 0, 0, 0, 0, 0};
    __builtin_amdgcn_tensor_load_to_lds(g0, g1, z4, z4, z8, 0);
#else
    __builtin_amdgcn_tensor_load_to_lds(g0, g1, z4, z4, 0);
#endif
}
#endif

// ---------------------------------------------------------------------------
// Weight pack: src f32 [E][Osrc][Isrc][3][3] -> dst f16 [E][Opad][KP]
// ---------------------------------------------------------------------------
__global__ __launch_bounds__(256) void pack_w_kernel(
    const float* __restrict__ src, _Float16* __restrict__ dst,
    int E_, int Osrc, int Opad, int Isrc, int Cpad)
{
    const int KR = 9 * Cpad;
    const int KP = (KR + 31) & ~31;
    const long total = (long)E_ * Opad * KP;
    for (long t = (long)blockIdx.x * 256 + threadIdx.x; t < total;
         t += (long)gridDim.x * 256) {
        int  k = (int)(t % KP);
        long r = t / KP;
        int  o = (int)(r % Opad);
        int  e = (int)(r / Opad);
        float v = 0.0f;
        if (k < KR) {
            int kykx = k / Cpad;
            int c    = k % Cpad;
            if (o < Osrc && c < Isrc)
                v = src[((long)(e * Osrc + o) * Isrc + c) * 9 + kykx];
        }
        dst[t] = (_Float16)v;
    }
}

// ---------------------------------------------------------------------------
// Input pack: f32 [N][3][32][32] -> f16 HWC [N][32][32][32]
// ---------------------------------------------------------------------------
__global__ __launch_bounds__(256) void pack_in_kernel(
    const float* __restrict__ in, _Float16* __restrict__ dst, int total)
{
    int t = blockIdx.x * 256 + threadIdx.x;
    if (t >= total) return;
    int c = t & 31;
    int p = (t >> 5) & 1023;
    int n = t >> 15;
    dst[t] = (c < 3) ? (_Float16)in[(n * 3 + c) * 1024 + p] : (_Float16)0.0f;
}

// ---------------------------------------------------------------------------
// Generic 32x32 conv3x3 (SAME) as WMMA GEMM.  grid=(N, O/16), block=256.
// 4 phases of 8 output rows; 10x34xCIN B halo slab in LDS (async load);
// 16xKP A weight tile in LDS (TDM).
// ---------------------------------------------------------------------------
template <int CIN>
__global__ __launch_bounds__(256) void conv32_wmma(
    const _Float16* __restrict__ src,
    const _Float16* __restrict__ W,
    const float*    __restrict__ bias,
    const int*      __restrict__ labels,
    int O,
    const _Float16* __restrict__ skip,
    int relu,
    _Float16* __restrict__ dst)
{
    constexpr int KR = 9 * CIN;
    constexpr int KP = (KR + 31) & ~31;
    __shared__ _Float16 smA[16 * KP];
    __shared__ _Float16 smB[10 * 34 * CIN];

    const int sample = blockIdx.x;
    const int mtile  = blockIdx.y;
    const int lbl    = labels[sample];

    const _Float16* Wb    = W + ((size_t)lbl * O + (size_t)mtile * 16) * KP;
    const float*    bb    = bias + (size_t)lbl * O + mtile * 16;
    const _Float16* sbase = src + (size_t)sample * 1024 * CIN;

    const int tid  = threadIdx.x;
    const int lane = tid & 31;
    const int wv   = tid >> 5;
    const int hi   = lane >> 4;
    const int mrow = lane & 15;

    // ---- stage A tile (16 x KP weights) into LDS ----
#if USE_TDM
    if (wv == 0) {
        tdm_load_tile(Wb, lds_off_of(smA), KP, 16);
        __builtin_amdgcn_s_wait_tensorcnt(0);
    }
#else
    for (int t = tid; t < 16 * KP / 8; t += 256)
        ((half8*)smA)[t] = ((const half8*)Wb)[t];
#endif
    __syncthreads();

    const _Float16* Arow = smA + (size_t)mrow * KP;

    for (int phase = 0; phase < 4; ++phase) {
        // ---- stage 10x34xCIN halo slab (rows phase*8-1 .. phase*8+8) ----
        for (int pos = tid; pos < 10 * 34; pos += 256) {
            int ly = pos / 34, lx = pos % 34;
            int gy = phase * 8 - 1 + ly;
            int gx = lx - 1;
            _Float16* drow = &smB[(ly * 34 + lx) * CIN];
            if (gy < 0 || gy > 31 || gx < 0 || gx > 31) {
#pragma unroll
                for (int c = 0; c < CIN; c += 8) {
                    half8 z;
#pragma unroll
                    for (int i = 0; i < 8; ++i) z[i] = (_Float16)0.0f;
                    *(half8*)&drow[c] = z;
                }
            } else {
                const _Float16* srow = sbase + (size_t)(gy * 32 + gx) * CIN;
#pragma unroll
                for (int c = 0; c < CIN; c += 8) {
#if USE_ASYNC_LDS
                    __builtin_amdgcn_global_load_async_to_lds_b128(
                        as1p(srow + c), as3p(drow + c), 0, 0);
#else
                    *(half8*)&drow[c] = *(const half8*)&srow[c];
#endif
                }
            }
        }
#if USE_ASYNC_LDS
        __builtin_amdgcn_s_wait_asynccnt(0);
#endif
        __syncthreads();

        // ---- compute: 256 pixels = 16 n-tiles, 2 per wave ----
        for (int t = 0; t < 2; ++t) {
            const int nt  = wv * 2 + t;
            const int nn  = nt * 16 + (lane & 15);
            const int pyl = nn >> 5;     // 0..7 local row
            const int px  = nn & 31;

            float8 acc;
#pragma unroll
            for (int v = 0; v < 8; ++v) acc[v] = bb[v + 8 * hi];

            for (int k0 = 0; k0 < KP; k0 += 32) {
                half8  alo = *(const half8*)(Arow + k0 + 16 * hi);
                half8  ahi = *(const half8*)(Arow + k0 + 16 * hi + 8);
                half16 a   = SHUF16(alo, ahi);

                const int kb = k0 + 16 * hi;
                half16 b;
                if (kb < KR) {
                    const int kykx = kb / CIN;
                    const int c0   = kb % CIN;   // {0,16,32}: 16B aligned
                    const int ky   = kykx / 3;
                    const int kx   = kykx - ky * 3;
                    const _Float16* bp =
                        &smB[((pyl + ky) * 34 + (px + kx)) * CIN + c0];
                    b = SHUF16(*(const half8*)bp, *(const half8*)(bp + 8));
                } else {
                    b = h16_zero();
                }

                acc = __builtin_amdgcn_wmma_f32_16x16x32_f16(
                    false, a, false, b, (short)0, acc, false, false);
            }

            const int py = phase * 8 + pyl;
            const size_t dbase =
                ((size_t)sample * 1024 + (size_t)(py * 32 + px)) * O + mtile * 16;
            const _Float16* sk = skip ? (skip + dbase) : nullptr;
#pragma unroll
            for (int v = 0; v < 8; ++v) {
                int   o   = v + 8 * hi;
                float val = acc[v];
                if (sk)   val += (float)sk[o];
                if (relu) val = fmaxf(val, 0.0f);
                dst[dbase + o] = (_Float16)val;
            }
        }
        __syncthreads();
    }
}

// ---------------------------------------------------------------------------
// Tail conv3x3 on 128x128 with fused pixel-shuffle read of up-conv output.
// grid = (N, 16 spatial tiles of 32x32). O padded 3->16.
// ---------------------------------------------------------------------------
__global__ __launch_bounds__(256) void tail_wmma(
    const _Float16* __restrict__ upo,
    const _Float16* __restrict__ W,
    const float*    __restrict__ bias,
    const int*      __restrict__ labels,
    float* __restrict__ out)
{
    constexpr int CIN = 48, KR = 432, KP = 448;
    __shared__ _Float16 smA[16 * KP];
    __shared__ _Float16 smB[10 * 34 * CIN];

    const int sample = blockIdx.x;
    const int tile   = blockIdx.y;
    const int ty     = tile >> 2, tx = tile & 3;
    const int lbl    = labels[sample];

    const _Float16* Wb = W + (size_t)lbl * 16 * KP;
    const float*    bb = bias + lbl * 3;
    const _Float16* ub = upo + (size_t)sample * 1024 * 768;

    const int tid  = threadIdx.x;
    const int lane = tid & 31;
    const int wv   = tid >> 5;
    const int hi   = lane >> 4;
    const int mrow = lane & 15;

#if USE_TDM
    if (wv == 0) {
        tdm_load_tile(Wb, lds_off_of(smA), KP, 16);
        __builtin_amdgcn_s_wait_tensorcnt(0);
    }
#else
    for (int t = tid; t < 16 * KP / 8; t += 256)
        ((half8*)smA)[t] = ((const half8*)Wb)[t];
#endif
    __syncthreads();

    const _Float16* Arow = smA + (size_t)mrow * KP;

    for (int phase = 0; phase < 4; ++phase) {
        // ---- stage halo slab through the pixel-shuffle mapping ----
        for (int pos = tid; pos < 10 * 34; pos += 256) {
            int ly = pos / 34, lx = pos % 34;
            int gy = ty * 32 + phase * 8 - 1 + ly;   // global Y in 0..127
            int gx = tx * 32 - 1 + lx;               // global X
            _Float16* drow = &smB[(ly * 34 + lx) * CIN];
            if (gy < 0 || gy > 127 || gx < 0 || gx > 127) {
                for (int c = 0; c < CIN; ++c) drow[c] = (_Float16)0.0f;
            } else {
                const size_t base =
                    (size_t)((gy >> 2) * 32 + (gx >> 2)) * 768 +
                    ((gy & 3) << 2) + (gx & 3);
                for (int c = 0; c < CIN; ++c)
                    drow[c] = ub[base + (size_t)c * 16];
            }
        }
        __syncthreads();

        for (int t = 0; t < 2; ++t) {
            const int nt  = wv * 2 + t;
            const int nn  = nt * 16 + (lane & 15);
            const int pyl = nn >> 5;
            const int px  = nn & 31;

            float8 acc;
#pragma unroll
            for (int v = 0; v < 8; ++v) {
                int o = v + 8 * hi;
                acc[v] = (o < 3) ? bb[o] : 0.0f;
            }

            for (int k0 = 0; k0 < KP; k0 += 32) {
                half8  alo = *(const half8*)(Arow + k0 + 16 * hi);
                half8  ahi = *(const half8*)(Arow + k0 + 16 * hi + 8);
                half16 a   = SHUF16(alo, ahi);

                const int kb = k0 + 16 * hi;
                half16 b;
                if (kb < KR) {
                    const int kykx = kb / CIN;
                    const int c0   = kb % CIN;
                    const int ky   = kykx / 3;
                    const int kx   = kykx - ky * 3;
                    const _Float16* bp =
                        &smB[((pyl + ky) * 34 + (px + kx)) * CIN + c0];
                    b = SHUF16(*(const half8*)bp, *(const half8*)(bp + 8));
                } else {
                    b = h16_zero();
                }

                acc = __builtin_amdgcn_wmma_f32_16x16x32_f16(
                    false, a, false, b, (short)0, acc, false, false);
            }

            const int Y = ty * 32 + phase * 8 + pyl;
            const int X = tx * 32 + px;
#pragma unroll
            for (int v = 0; v < 8; ++v) {
                int o = v + 8 * hi;
                if (o < 3)
                    out[(((size_t)sample * 3 + o) << 14) + (Y << 7) + X] = acc[v];
            }
        }
        __syncthreads();
    }
}

// ---------------------------------------------------------------------------
// Host launcher
// ---------------------------------------------------------------------------
extern "C" void kernel_launch(void* const* d_in, const int* in_sizes, int n_in,
                              void* d_out, int out_size, void* d_ws, size_t ws_size,
                              hipStream_t stream) {
    (void)in_sizes; (void)n_in; (void)out_size; (void)ws_size;

    constexpr int N = 128, E = 10;

    const float* inputs = (const float*)d_in[0];
    const int*   labels = (const int*)d_in[1];
    const float* head_w = (const float*)d_in[2];
    const float* head_b = (const float*)d_in[3];
    const float* rb1_w  = (const float*)d_in[4];
    const float* rb1_b  = (const float*)d_in[5];
    const float* rb2_w  = (const float*)d_in[6];
    const float* rb2_b  = (const float*)d_in[7];
    const float* body_w = (const float*)d_in[8];
    const float* body_b = (const float*)d_in[9];
    const float* up_w   = (const float*)d_in[10];
    const float* up_b   = (const float*)d_in[11];
    const float* tail_w = (const float*)d_in[12];
    const float* tail_b = (const float*)d_in[13];
    float* out = (float*)d_out;

    size_t off = 0;
    auto alloc = [&](size_t halves) -> _Float16* {
        _Float16* p = (_Float16*)((char*)d_ws + off);
        off += ((halves * 2 + 255) & ~(size_t)255);
        return p;
    };
    _Float16* wHead = alloc((size_t)E * 48 * 288);
    _Float16* wRb1  = alloc((size_t)E * 48 * 448);
    _Float16* wRb2  = alloc((size_t)E * 48 * 448);
    _Float16* wBody = alloc((size_t)E * 48 * 448);
    _Float16* wUp   = alloc((size_t)E * 768 * 448);
    _Float16* wTail = alloc((size_t)E * 16 * 448);
    _Float16* actIn = alloc((size_t)N * 1024 * 32);
    _Float16* f0    = alloc((size_t)N * 1024 * 48);
    _Float16* t1    = alloc((size_t)N * 1024 * 48);
    _Float16* fsum  = alloc((size_t)N * 1024 * 48);
    _Float16* fb    = alloc((size_t)N * 1024 * 48);
    _Float16* upo   = alloc((size_t)N * 1024 * 768);

    auto gw = [](long total) { return dim3((unsigned)((total + 255) / 256)); };

    pack_w_kernel<<<gw((long)E * 48 * 288), 256, 0, stream>>>(head_w, wHead, E, 48, 48, 3, 32);
    pack_w_kernel<<<gw((long)E * 48 * 448), 256, 0, stream>>>(rb1_w, wRb1, E, 48, 48, 48, 48);
    pack_w_kernel<<<gw((long)E * 48 * 448), 256, 0, stream>>>(rb2_w, wRb2, E, 48, 48, 48, 48);
    pack_w_kernel<<<gw((long)E * 48 * 448), 256, 0, stream>>>(body_w, wBody, E, 48, 48, 48, 48);
    pack_w_kernel<<<gw((long)E * 768 * 448), 256, 0, stream>>>(up_w, wUp, E, 768, 768, 48, 48);
    pack_w_kernel<<<gw((long)E * 16 * 448), 256, 0, stream>>>(tail_w, wTail, E, 3, 16, 48, 48);

    pack_in_kernel<<<gw((long)N * 1024 * 32), 256, 0, stream>>>(inputs, actIn, N * 1024 * 32);

    conv32_wmma<32><<<dim3(N, 3), 256, 0, stream>>>(
        actIn, wHead, head_b, labels, 48, nullptr, 0, f0);
    conv32_wmma<48><<<dim3(N, 3), 256, 0, stream>>>(
        f0, wRb1, rb1_b, labels, 48, nullptr, 1, t1);
    conv32_wmma<48><<<dim3(N, 3), 256, 0, stream>>>(
        t1, wRb2, rb2_b, labels, 48, f0, 0, fsum);
    conv32_wmma<48><<<dim3(N, 3), 256, 0, stream>>>(
        fsum, wBody, body_b, labels, 48, f0, 0, fb);
    conv32_wmma<48><<<dim3(N, 48), 256, 0, stream>>>(
        fb, wUp, up_b, labels, 768, nullptr, 0, upo);
    tail_wmma<<<dim3(N, 16), 256, 0, stream>>>(upo, wTail, tail_b, labels, out);
}
